// NonsharedModel_76819785056656
// MI455X (gfx1250) — matrix operand order repaired and checked by reading, not compiled, not run on hardware
//
#include <hip/hip_runtime.h>

typedef __attribute__((ext_vector_type(16))) _Float16 v16h;
typedef __attribute__((ext_vector_type(8)))  _Float16 v8h;
typedef __attribute__((ext_vector_type(8)))  float    v8f;
typedef __attribute__((ext_vector_type(4)))  int      v4i;

#define DIM     17
#define TSTEPS  204
#define BATCH   16384
#define HID     100
#define T1      205        // T+1 (x last-dim)
#define BM      128        // batch rows per workgroup
#define NTHREADS 256       // 8 waves (wave32)
#define KP      128        // padded K for hidden layers (100 -> 128)
#define NP      112        // padded N (100 -> 7 tiles of 16)
#define NCHUNK  8          // timestep chunks for the z-kernel
#define NNETS   205        // 204 z-nets + 1 y-net (index 204)

static constexpr float DT_F  = 17.0f / 204.0f;
static constexpr float EPS_F = 1e-5f;

// ---------------------------------------------------------------------------
// CDNA5 async global->LDS path (guarded; falls back to load+ds_store).
#if __has_builtin(__builtin_amdgcn_global_load_async_to_lds_b128)
#define HAVE_ASYNC_LDS 1
#else
#define HAVE_ASYNC_LDS 0
#endif

__device__ inline void wait_asynccnt0() {
#if HAVE_ASYNC_LDS
#if __has_builtin(__builtin_amdgcn_s_wait_asynccnt)
  __builtin_amdgcn_s_wait_asynccnt(0);
#else
  asm volatile("s_wait_asynccnt 0" ::: "memory");
#endif
#endif
}

// Stage one padded f16 weight tile [NP][KP] (28672 B) global -> LDS.
// 1792 chunks of 16 B -> 7 chunks per thread.
__device__ inline void stage_weights(const _Float16* __restrict__ g,
                                     _Float16* s, int tid) {
#if HAVE_ASYNC_LDS
#pragma unroll
  for (int i = 0; i < (NP * KP) / (8 * NTHREADS); ++i) {  // 7
    const int c = tid + i * NTHREADS;
    __builtin_amdgcn_global_load_async_to_lds_b128(
        (__attribute__((address_space(1))) v4i*)(g + c * 8),
        (__attribute__((address_space(3))) v4i*)(s + c * 8),
        0, 0);
  }
#else
#pragma unroll
  for (int i = 0; i < (NP * KP) / (8 * NTHREADS); ++i) {
    const int c = tid + i * NTHREADS;
    *(v8h*)(s + c * 8) = *(const v8h*)(g + c * 8);
  }
#endif
}

// ---------------------------------------------------------------------------
__device__ inline v8f vzero8() {
  v8f v;
#pragma unroll
  for (int i = 0; i < 8; ++i) v[i] = 0.0f;
  return v;
}

__device__ inline v16h cat16(v8h lo, v8h hi) {
  return __builtin_shufflevector(lo, hi, 0,1,2,3,4,5,6,7,8,9,10,11,12,13,14,15);
}

// One hidden layer: D = relu(A @ W^T + bias).  A: [BM][KP] f16 LDS (wave owns
// its 16 rows), W: [NP][KP] f16 LDS; bias: NP floats (LDS or global).
__device__ inline void mlp_layer(const _Float16* __restrict__ sA,
                                 const _Float16* __restrict__ sW,
                                 const float* __restrict__ bias,
                                 _Float16* __restrict__ sOut,
                                 int kSteps, int lane, int rowBase) {
  const int nm = lane & 15;            // A row / B column within tile
  const int ah = (lane >> 4) * 8;      // A: K-half select (ISA 16-bit A layout)
  const int bh = (lane >> 4) * 8;      // B: K-half select (same K order as A)
  v8f acc[7];
#pragma unroll
  for (int nt = 0; nt < 7; ++nt) acc[nt] = vzero8();

  for (int kt = 0; kt < kSteps; ++kt) {
    const int kb = kt * 32;
    const _Float16* ap = sA + (rowBase + nm) * KP + kb + ah;
    v16h a = cat16(*(const v8h*)ap, *(const v8h*)(ap + 16));
#pragma unroll
    for (int nt = 0; nt < 7; ++nt) {
      const _Float16* bp = sW + (nt * 16 + nm) * KP + kb + bh;
      v16h b = cat16(*(const v8h*)bp, *(const v8h*)(bp + 16));
      acc[nt] = __builtin_amdgcn_wmma_f32_16x16x32_f16(
          false, a, false, b, (short)0, acc[nt], false, false);
    }
  }
  const int mrow = rowBase + 8 * (lane >> 4);   // C/D layout: M = r + 8*(lane/16)
#pragma unroll
  for (int nt = 0; nt < 7; ++nt) {
    float bv = bias[nt * 16 + nm];
#pragma unroll
    for (int r = 0; r < 8; ++r) {
      float v = acc[nt][r] + bv;
      v = v > 0.0f ? v : 0.0f;
      sOut[(mrow + r) * KP + nt * 16 + nm] = (_Float16)v;
    }
  }
}

// Final layer (no relu), NT output tiles, f32 result into sZ [BM][32].
template <int NT>
__device__ inline void mlp_final(const _Float16* __restrict__ sA,
                                 const _Float16* __restrict__ sW,
                                 const float* __restrict__ bias,
                                 float* __restrict__ sZ,
                                 int lane, int rowBase) {
  const int nm = lane & 15;
  const int ah = (lane >> 4) * 8;
  const int bh = (lane >> 4) * 8;
  v8f acc[NT];
#pragma unroll
  for (int nt = 0; nt < NT; ++nt) acc[nt] = vzero8();

  for (int kt = 0; kt < 4; ++kt) {           // K = 100 -> 128
    const int kb = kt * 32;
    const _Float16* ap = sA + (rowBase + nm) * KP + kb + ah;
    v16h a = cat16(*(const v8h*)ap, *(const v8h*)(ap + 16));
#pragma unroll
    for (int nt = 0; nt < NT; ++nt) {
      const _Float16* bp = sW + (nt * 16 + nm) * KP + kb + bh;
      v16h b = cat16(*(const v8h*)bp, *(const v8h*)(bp + 16));
      acc[nt] = __builtin_amdgcn_wmma_f32_16x16x32_f16(
          false, a, false, b, (short)0, acc[nt], false, false);
    }
  }
  const int mrow = rowBase + 8 * (lane >> 4);
#pragma unroll
  for (int nt = 0; nt < NT; ++nt) {
    float bv = bias[nt * 16 + nm];
#pragma unroll
    for (int r = 0; r < 8; ++r) {
      sZ[(mrow + r) * 32 + nt * 16 + nm] = acc[nt][r] + bv;
    }
  }
}

// ---------------------------------------------------------------------------
// Batch-statistics kernel: deterministic tree reductions per (t, d).
__global__ __launch_bounds__(NTHREADS)
void stats_kernel(const float* __restrict__ x,
                  float* __restrict__ mean, float* __restrict__ var) {
  __shared__ float red[NTHREADS];
  __shared__ float smean;
  const int t = blockIdx.x;
  const int tid = threadIdx.x;
  float s[DIM], q[DIM];
#pragma unroll
  for (int d = 0; d < DIM; ++d) { s[d] = 0.0f; q[d] = 0.0f; }
  for (int b = tid; b < BATCH; b += NTHREADS) {
#pragma unroll
    for (int d = 0; d < DIM; ++d) {
      float v = x[((size_t)b * DIM + d) * T1 + t];
      s[d] += v; q[d] += v * v;
    }
  }
  for (int d = 0; d < DIM; ++d) {
    red[tid] = s[d]; __syncthreads();
    for (int off = NTHREADS / 2; off > 0; off >>= 1) {
      if (tid < off) red[tid] += red[tid + off];
      __syncthreads();
    }
    if (tid == 0) { smean = red[0] * (1.0f / BATCH); mean[t * DIM + d] = smean; }
    __syncthreads();
    red[tid] = q[d]; __syncthreads();
    for (int off = NTHREADS / 2; off > 0; off >>= 1) {
      if (tid < off) red[tid] += red[tid + off];
      __syncthreads();
    }
    if (tid == 0) {
      float m = smean;
      var[t * DIM + d] = red[0] * (1.0f / BATCH) - m * m;
    }
    __syncthreads();
  }
}

// ---------------------------------------------------------------------------
// Pre-convert all weights into a padded f16 table wtab[net][layer][NP][KP]
// (zero-padded so padded tiles contribute exactly 0) + padded f32 biases.
__global__ __launch_bounds__(NTHREADS)
void convert_weights_kernel(
    const float* __restrict__ zW1, const float* __restrict__ zb1,
    const float* __restrict__ zW2, const float* __restrict__ zb2,
    const float* __restrict__ zW3, const float* __restrict__ zb3,
    const float* __restrict__ zW4, const float* __restrict__ zb4,
    const float* __restrict__ zW5, const float* __restrict__ zb5,
    const float* __restrict__ yW1, const float* __restrict__ yb1,
    const float* __restrict__ yW2, const float* __restrict__ yb2,
    const float* __restrict__ yW3, const float* __restrict__ yb3,
    const float* __restrict__ yW4, const float* __restrict__ yb4,
    const float* __restrict__ yW5, const float* __restrict__ yb5,
    _Float16* __restrict__ wtab, float* __restrict__ btab) {
  const int net = blockIdx.x;            // 0..203 = z-nets, 204 = y-net
  const int l   = blockIdx.y;            // 0..4
  const int tid = threadIdx.x;
  const bool isY = (net == TSTEPS);

  const float* W; const float* bsrc; int nOut, nIn;
  if (l == 0) {
    nOut = HID; nIn = DIM;
    W = isY ? yW1 : zW1 + (size_t)net * HID * DIM;
    bsrc = isY ? yb1 : zb1 + (size_t)net * HID;
  } else if (l == 1) {
    nOut = HID; nIn = HID;
    W = isY ? yW2 : zW2 + (size_t)net * HID * HID;
    bsrc = isY ? yb2 : zb2 + (size_t)net * HID;
  } else if (l == 2) {
    nOut = HID; nIn = HID;
    W = isY ? yW3 : zW3 + (size_t)net * HID * HID;
    bsrc = isY ? yb3 : zb3 + (size_t)net * HID;
  } else if (l == 3) {
    nOut = HID; nIn = HID;
    W = isY ? yW4 : zW4 + (size_t)net * HID * HID;
    bsrc = isY ? yb4 : zb4 + (size_t)net * HID;
  } else {
    nOut = isY ? 1 : DIM; nIn = HID;
    W = isY ? yW5 : zW5 + (size_t)net * DIM * HID;
    bsrc = isY ? yb5 : zb5 + (size_t)net * DIM;
  }

  _Float16* dst = wtab + ((size_t)net * 5 + l) * (NP * KP);
  for (int idx = tid; idx < NP * KP; idx += NTHREADS) {
    const int n = idx >> 7;
    const int k = idx & (KP - 1);
    float v = (n < nOut && k < nIn) ? W[(size_t)n * nIn + k] : 0.0f;
    dst[idx] = (_Float16)v;
  }
  float* bdst = btab + ((size_t)net * 5 + l) * NP;
  for (int n = tid; n < NP; n += NTHREADS) bdst[n] = (n < nOut) ? bsrc[n] : 0.0f;
}

// ---------------------------------------------------------------------------
// FAST PATH: fused BN + 5-layer MLP using pre-converted f16 weights with
// async-DMA double-buffered weight staging (stage layer l+1 while computing l).
template <bool YNET>
__global__ __launch_bounds__(NTHREADS)
void bsde_mlp_fast(const float* __restrict__ x,  const float* __restrict__ dw,
                   const float* __restrict__ lambd,
                   const float* __restrict__ gamma, const float* __restrict__ beta,
                   const _Float16* __restrict__ wtab, const float* __restrict__ btab,
                   const float* __restrict__ meanv, const float* __restrict__ varv,
                   float* __restrict__ outv) {
  __shared__ __align__(16) _Float16 sH0[BM * KP];
  __shared__ __align__(16) _Float16 sH1[BM * KP];
  __shared__ __align__(16) _Float16 sWd[2][NP * KP];
  __shared__ float sZ[BM * 32];
  __shared__ float sNorm[4 * DIM];   // mean, rstd, gamma, beta
  __shared__ float sFac[DIM];

  const int tid = threadIdx.x;
  const int lane = tid & 31;
  const int wave = tid >> 5;
  const int rowBase = wave * 16;
  const int bBase = blockIdx.x * BM;

  // Zero activation buffers once: pad columns stay finite/zero forever.
  for (int i = tid; i < BM * KP; i += NTHREADS) {
    sH0[i] = (_Float16)0.0f;
    sH1[i] = (_Float16)0.0f;
  }

  int t0, t1;
  if (YNET) { t0 = 0; t1 = 1; }
  else {
    const int per = (TSTEPS + NCHUNK - 1) / NCHUNK;   // 26
    t0 = blockIdx.y * per;
    t1 = t0 + per; if (t1 > TSTEPS) t1 = TSTEPS;
  }
  float part = 0.0f;

  for (int t = t0; t < t1; ++t) {
    __syncthreads();  // prior iteration's sZ/sFac reads done before overwrite
    if (tid < DIM) {
      const int d = tid;
      sNorm[d]           = meanv[t * DIM + d];
      sNorm[DIM + d]     = rsqrtf(varv[t * DIM + d] + EPS_F);
      sNorm[2 * DIM + d] = YNET ? gamma[d] : gamma[t * DIM + d];
      sNorm[3 * DIM + d] = YNET ? beta[d]  : beta[t * DIM + d];
      if (!YNET) sFac[d] = (t < TSTEPS - 1) ? sqrtf(2.0f * lambd[d * TSTEPS + t]) : 1.0f;
    }
    __syncthreads();

    // BatchNorm input tile -> sH0 (f16, K columns 0..16)
    for (int idx = tid; idx < BM * DIM; idx += NTHREADS) {
      const int row = idx / DIM;
      const int d = idx - row * DIM;
      float xv = x[((size_t)(bBase + row) * DIM + d) * T1 + t];
      float h = (xv - sNorm[d]) * sNorm[DIM + d] * sNorm[2 * DIM + d] + sNorm[3 * DIM + d];
      sH0[row * KP + d] = (_Float16)h;
    }

    const int net = YNET ? TSTEPS : t;
    const _Float16* wl = wtab + (size_t)net * 5 * (NP * KP);
    const float*    bl = btab + (size_t)net * 5 * NP;

    stage_weights(wl + 0 * (NP * KP), sWd[0], tid);   // layer 1 weights
    wait_asynccnt0(); __syncthreads();                // weights + BN tile ready

    stage_weights(wl + 1 * (NP * KP), sWd[1], tid);   // prefetch layer 2
    mlp_layer(sH0, sWd[0], bl + 0 * NP, sH1, /*kSteps=*/1, lane, rowBase);
    wait_asynccnt0(); __syncthreads();

    stage_weights(wl + 2 * (NP * KP), sWd[0], tid);   // prefetch layer 3
    mlp_layer(sH1, sWd[1], bl + 1 * NP, sH0, /*kSteps=*/4, lane, rowBase);
    wait_asynccnt0(); __syncthreads();

    stage_weights(wl + 3 * (NP * KP), sWd[1], tid);   // prefetch layer 4
    mlp_layer(sH0, sWd[0], bl + 2 * NP, sH1, /*kSteps=*/4, lane, rowBase);
    wait_asynccnt0(); __syncthreads();

    stage_weights(wl + 4 * (NP * KP), sWd[0], tid);   // prefetch layer 5
    mlp_layer(sH1, sWd[1], bl + 3 * NP, sH0, /*kSteps=*/4, lane, rowBase);
    wait_asynccnt0(); __syncthreads();

    mlp_final<(YNET ? 1 : 2)>(sH0, sWd[0], bl + 4 * NP, sZ, lane, rowBase);
    __syncthreads();

    if (YNET) {
      if (tid < BM) outv[bBase + tid] = sZ[tid * 32];
    } else if (tid < BM) {
      const int b = bBase + tid;
      float c = 0.0f;
#pragma unroll
      for (int d = 0; d < DIM; ++d) {
        float z = sZ[tid * 32 + d];
        float dwv = dw[((size_t)b * DIM + d) * TSTEPS + t];
        c += DT_F * z * z + z * sFac[d] * dwv;
      }
      part += c;
    }
  }
  if (!YNET && tid < BM) {
    outv[(size_t)blockIdx.y * BATCH + bBase + tid] = part;
  }
}

// ---------------------------------------------------------------------------
// FALLBACK PATH (small workspace): convert f32 weights to LDS f16 per layer.
__device__ inline void load_weights(const float* __restrict__ Wg,
                                    const float* __restrict__ bg,
                                    int nOut, int nIn,
                                    _Float16* sW, float* sBias, int tid) {
  for (int idx = tid; idx < NP * KP; idx += NTHREADS) {
    int n = idx >> 7;
    int k = idx & (KP - 1);
    float v = (n < nOut && k < nIn) ? Wg[(size_t)n * nIn + k] : 0.0f;
    sW[idx] = (_Float16)v;
  }
  for (int n = tid; n < NP; n += NTHREADS) sBias[n] = (n < nOut) ? bg[n] : 0.0f;
}

template <bool YNET>
__global__ __launch_bounds__(NTHREADS)
void bsde_mlp_kernel(const float* __restrict__ x,  const float* __restrict__ dw,
                     const float* __restrict__ lambd,
                     const float* __restrict__ gamma, const float* __restrict__ beta,
                     const float* __restrict__ W1, const float* __restrict__ b1,
                     const float* __restrict__ W2, const float* __restrict__ b2,
                     const float* __restrict__ W3, const float* __restrict__ b3,
                     const float* __restrict__ W4, const float* __restrict__ b4,
                     const float* __restrict__ W5, const float* __restrict__ b5,
                     const float* __restrict__ meanv, const float* __restrict__ varv,
                     float* __restrict__ outv) {
  __shared__ __align__(16) _Float16 sH0[BM * KP];
  __shared__ __align__(16) _Float16 sH1[BM * KP];
  __shared__ __align__(16) _Float16 sW[NP * KP];
  __shared__ float    sBias[NP];
  __shared__ float    sZ[BM * 32];
  __shared__ float    sNorm[4 * DIM];
  __shared__ float    sFac[DIM];

  const int tid = threadIdx.x;
  const int lane = tid & 31;
  const int wave = tid >> 5;
  const int rowBase = wave * 16;
  const int bBase = blockIdx.x * BM;

  for (int i = tid; i < BM * KP; i += NTHREADS) {
    sH0[i] = (_Float16)0.0f;
    sH1[i] = (_Float16)0.0f;
  }

  int t0, t1;
  if (YNET) { t0 = 0; t1 = 1; }
  else {
    const int per = (TSTEPS + NCHUNK - 1) / NCHUNK;
    t0 = blockIdx.y * per;
    t1 = t0 + per; if (t1 > TSTEPS) t1 = TSTEPS;
  }
  float part = 0.0f;

  for (int t = t0; t < t1; ++t) {
    __syncthreads();
    if (tid < DIM) {
      const int d = tid;
      sNorm[d]           = meanv[t * DIM + d];
      sNorm[DIM + d]     = rsqrtf(varv[t * DIM + d] + EPS_F);
      sNorm[2 * DIM + d] = YNET ? gamma[d] : gamma[t * DIM + d];
      sNorm[3 * DIM + d] = YNET ? beta[d]  : beta[t * DIM + d];
      if (!YNET) sFac[d] = (t < TSTEPS - 1) ? sqrtf(2.0f * lambd[d * TSTEPS + t]) : 1.0f;
    }
    __syncthreads();

    for (int idx = tid; idx < BM * DIM; idx += NTHREADS) {
      const int row = idx / DIM;
      const int d = idx - row * DIM;
      float xv = x[((size_t)(bBase + row) * DIM + d) * T1 + t];
      float h = (xv - sNorm[d]) * sNorm[DIM + d] * sNorm[2 * DIM + d] + sNorm[3 * DIM + d];
      sH0[row * KP + d] = (_Float16)h;
    }

    const int tw = YNET ? 0 : t;
    load_weights(W1 + (size_t)tw * HID * DIM, b1 + (size_t)tw * HID, HID, DIM, sW, sBias, tid);
    __syncthreads();
    mlp_layer(sH0, sW, sBias, sH1, 1, lane, rowBase);
    __syncthreads();

    load_weights(W2 + (size_t)tw * HID * HID, b2 + (size_t)tw * HID, HID, HID, sW, sBias, tid);
    __syncthreads();
    mlp_layer(sH1, sW, sBias, sH0, 4, lane, rowBase);
    __syncthreads();

    load_weights(W3 + (size_t)tw * HID * HID, b3 + (size_t)tw * HID, HID, HID, sW, sBias, tid);
    __syncthreads();
    mlp_layer(sH0, sW, sBias, sH1, 4, lane, rowBase);
    __syncthreads();

    load_weights(W4 + (size_t)tw * HID * HID, b4 + (size_t)tw * HID, HID, HID, sW, sBias, tid);
    __syncthreads();
    mlp_layer(sH1, sW, sBias, sH0, 4, lane, rowBase);
    __syncthreads();

    const int nOut = YNET ? 1 : DIM;
    load_weights(W5 + (size_t)tw * nOut * HID, b5 + (size_t)tw * nOut, nOut, HID, sW, sBias, tid);
    __syncthreads();
    mlp_final<(YNET ? 1 : 2)>(sH0, sW, sBias, sZ, lane, rowBase);
    __syncthreads();

    if (YNET) {
      if (tid < BM) outv[bBase + tid] = sZ[tid * 32];
    } else if (tid < BM) {
      const int b = bBase + tid;
      float c = 0.0f;
#pragma unroll
      for (int d = 0; d < DIM; ++d) {
        float z = sZ[tid * 32 + d];
        float dwv = dw[((size_t)b * DIM + d) * TSTEPS + t];
        c += DT_F * z * z + z * sFac[d] * dwv;
      }
      part += c;
    }
  }
  if (!YNET && tid < BM) {
    outv[(size_t)blockIdx.y * BATCH + bBase + tid] = part;
  }
}

// ---------------------------------------------------------------------------
__global__ __launch_bounds__(NTHREADS)
void finalize_kernel(const float* __restrict__ y0,
                     const float* __restrict__ part,
                     float* __restrict__ out) {
  const int b = blockIdx.x * NTHREADS + threadIdx.x;
  if (b < BATCH) {
    float v = y0[b];
#pragma unroll
    for (int ch = 0; ch < NCHUNK; ++ch) v += part[(size_t)ch * BATCH + b];
    out[b] = v;
  }
}

// ---------------------------------------------------------------------------
extern "C" void kernel_launch(void* const* d_in, const int* in_sizes, int n_in,
                              void* d_out, int out_size, void* d_ws, size_t ws_size,
                              hipStream_t stream) {
  const float* dw      = (const float*)d_in[0];
  const float* x       = (const float*)d_in[1];
  const float* lambd   = (const float*)d_in[2];
  const float* z_gamma = (const float*)d_in[3];
  const float* z_beta  = (const float*)d_in[4];
  const float* zW1 = (const float*)d_in[5];  const float* zb1 = (const float*)d_in[6];
  const float* zW2 = (const float*)d_in[7];  const float* zb2 = (const float*)d_in[8];
  const float* zW3 = (const float*)d_in[9];  const float* zb3 = (const float*)d_in[10];
  const float* zW4 = (const float*)d_in[11]; const float* zb4 = (const float*)d_in[12];
  const float* zW5 = (const float*)d_in[13]; const float* zb5 = (const float*)d_in[14];
  const float* y_gamma = (const float*)d_in[15];
  const float* y_beta  = (const float*)d_in[16];
  const float* yW1 = (const float*)d_in[17]; const float* yb1 = (const float*)d_in[18];
  const float* yW2 = (const float*)d_in[19]; const float* yb2 = (const float*)d_in[20];
  const float* yW3 = (const float*)d_in[21]; const float* yb3 = (const float*)d_in[22];
  const float* yW4 = (const float*)d_in[23]; const float* yb4 = (const float*)d_in[24];
  const float* yW5 = (const float*)d_in[25]; const float* yb5 = (const float*)d_in[26];

  // Workspace layout:
  //   floats: mean[4096] | var[4096] | y0[B] | partial[NCHUNK*B] | btab[NNETS*5*NP]
  //   then 16B-aligned: wtab f16 [NNETS*5*NP*KP]
  float* ws   = (float*)d_ws;
  float* mean = ws;
  float* var  = ws + 4096;
  float* y0   = ws + 8192;
  float* part = ws + 8192 + BATCH;
  float* btab = part + (size_t)NCHUNK * BATCH;
  size_t btab_elems = (size_t)NNETS * 5 * NP;
  size_t wtab_off_f = (8192 + BATCH + (size_t)NCHUNK * BATCH + btab_elems + 3) & ~(size_t)3;
  _Float16* wtab = (_Float16*)(ws + wtab_off_f);
  size_t need = wtab_off_f * 4 + (size_t)NNETS * 5 * NP * KP * 2;

  stats_kernel<<<TSTEPS, NTHREADS, 0, stream>>>(x, mean, var);

  if (ws_size >= need) {
    // Fast path: pre-converted padded f16 weights + async double-buffered LDS.
    convert_weights_kernel<<<dim3(NNETS, 5), NTHREADS, 0, stream>>>(
        zW1, zb1, zW2, zb2, zW3, zb3, zW4, zb4, zW5, zb5,
        yW1, yb1, yW2, yb2, yW3, yb3, yW4, yb4, yW5, yb5,
        wtab, btab);

    bsde_mlp_fast<true><<<dim3(BATCH / BM), NTHREADS, 0, stream>>>(
        x, nullptr, nullptr, y_gamma, y_beta, wtab, btab, mean, var, y0);

    bsde_mlp_fast<false><<<dim3(BATCH / BM, NCHUNK), NTHREADS, 0, stream>>>(
        x, dw, lambd, z_gamma, z_beta, wtab, btab, mean, var, part);
  } else {
    // Fallback: per-layer f32->f16 conversion into LDS.
    bsde_mlp_kernel<true><<<dim3(BATCH / BM), NTHREADS, 0, stream>>>(
        x, nullptr, nullptr, y_gamma, y_beta,
        yW1, yb1, yW2, yb2, yW3, yb3, yW4, yb4, yW5, yb5,
        mean, var, y0);

    bsde_mlp_kernel<false><<<dim3(BATCH / BM, NCHUNK), NTHREADS, 0, stream>>>(
        x, dw, lambd, z_gamma, z_beta,
        zW1, zb1, zW2, zb2, zW3, zb3, zW4, zb4, zW5, zb5,
        mean, var, part);
  }

  finalize_kernel<<<(BATCH + NTHREADS - 1) / NTHREADS, NTHREADS, 0, stream>>>(
      y0, part, (float*)d_out);
}
